// MSA_57372173140045
// MI455X (gfx1250) — matrix operand order repaired and checked by reading, not compiled, not run on hardware
//
#include <hip/hip_runtime.h>
#include <hip/hip_bf16.h>

// ---------------------------------------------------------------------------
// MSA (q=k=v=proj_q(x) source bug preserved) for MI455X / gfx1250, wave32.
//   B=2, N=2048, D=1024, H=16, d=64.  ~78 GFLOP, compute-bound ->
//   v_wmma_f32_16x16x32_f16 for attention + out-proj, flash-style online
//   softmax (log2 domain), row-sums via an extra ones-operand WMMA,
//   double-buffered LDS key tiles with global prefetch.
// ---------------------------------------------------------------------------

typedef __attribute__((ext_vector_type(16))) _Float16 v16h;
typedef __attribute__((ext_vector_type(8)))  _Float16 v8h;
typedef __attribute__((ext_vector_type(4)))  _Float16 v4h;
typedef __attribute__((ext_vector_type(8)))  float    v8f;

#define NQ   2048   // sequence length
#define DH   64     // head dim
#define NH   16     // heads
#define DIM  1024
#define BSZ  2
#define QT   128    // query rows per block (8 waves x 16)
#define KT   32     // key tile

__device__ __forceinline__ v8f wmma_f16(v16h a, v16h b, v8f c) {
  // v_wmma_f32_16x16x32_f16: D = A(16x32 f16) * B(32x16 f16) + C(16x16 f32)
  return __builtin_amdgcn_wmma_f32_16x16x32_f16(false, a, false, b,
                                                (short)0, c, false, false);
}

// A fragment (16x32 f16). base -> element (row0, k0); ldm = row stride (halves).
// ISA layout: lane L holds row L&15; K base = (L<16 ? 0 : 8);
// halves 0..7 -> K=kb..kb+7, halves 8..15 -> K=kb+16..kb+23 (contiguous runs).
__device__ __forceinline__ v16h load_a_frag(const _Float16* base, int ldm, int lane) {
  const int m  = lane & 15;
  const int kb = (lane & 16) >> 1;          // 0 or 8
  const _Float16* p = base + m * ldm + kb;
  union { v16h v; v8h h[2]; } u;
  u.h[0] = *(const v8h*)(p);
  u.h[1] = *(const v8h*)(p + 16);
  return u.v;
}

// B fragment (32x16 f16) from memory stored as [col][k] (ldn = col stride).
// ISA layout: lane L holds col L&15; K base = (L<16 ? 0 : 8); halves 0..7 -> K=kb..kb+7, 8..15 -> K=kb+16..kb+23.
__device__ __forceinline__ v16h load_b_frag(const _Float16* base, int ldn, int lane) {
  const int c  = lane & 15;
  const int kb = (lane & 16) >> 1;          // 0 or 8
  const _Float16* p = base + c * ldn + kb;
  union { v16h v; v8h h[2]; } u;
  u.h[0] = *(const v8h*)(p);
  u.h[1] = *(const v8h*)(p + 16);
  return u.v;
}

// Per-row max over a C-fragment (v8f): VGPR i holds row i (lanes 0-15) or
// row 8+i (lanes 16-31); butterfly across the 16-lane half (ds_bpermute).
__device__ __forceinline__ v8f rowmax8(v8f t) {
#pragma unroll
  for (int m = 1; m < 16; m <<= 1) {
#pragma unroll
    for (int i = 0; i < 8; ++i) t[i] = fmaxf(t[i], __shfl_xor(t[i], m, 32));
  }
  return t;
}

// ---------------------------------------------------------------------------
// Kernel 0: Wp f32 -> f16
// ---------------------------------------------------------------------------
__global__ __launch_bounds__(256) void wpconv_kernel(const float* __restrict__ wp,
                                                     _Float16* __restrict__ wph) {
  const int idx = blockIdx.x * 256 + threadIdx.x;     // over 262144 float4s
  const float4 v = ((const float4*)wp)[idx];
  v4h o;
  o[0] = (_Float16)v.x; o[1] = (_Float16)v.y;
  o[2] = (_Float16)v.z; o[3] = (_Float16)v.w;
  *(v4h*)(wph + (size_t)idx * 4) = o;
}

// ---------------------------------------------------------------------------
// Kernel 1: q[b,h,n,e] = sum_d x[b,n,h*64+d] * Wq[h,e,d]   -> f16 [BH][N][64]
// Tiny fraction of FLOPs; f32 VALU dot, one thread per output element.
// ---------------------------------------------------------------------------
__global__ __launch_bounds__(256) void qproj_kernel(const float* __restrict__ x,
                                                    const float* __restrict__ Wq,
                                                    _Float16* __restrict__ qh) {
  const int e  = threadIdx.x & 63;
  const int nn = threadIdx.x >> 6;
  const int n  = blockIdx.x * 4 + nn;
  const int bh = blockIdx.y;
  const int b  = bh >> 4, h = bh & 15;
  const float4* xr = (const float4*)(x + ((size_t)(b * NQ + n)) * DIM + h * DH);
  const float4* wr = (const float4*)(Wq + ((size_t)(h * DH + e)) * DH);
  float acc = 0.f;
#pragma unroll
  for (int i = 0; i < 16; ++i) {
    const float4 a = xr[i], w = wr[i];
    acc += a.x * w.x + a.y * w.y + a.z * w.z + a.w * w.w;
  }
  qh[((size_t)bh * NQ + n) * DH + e] = (_Float16)acc;
}

// ---------------------------------------------------------------------------
// Kernel 2: flash attention per (b,h).  S_log2 = (8*log2 e) * Q Q^T (scale
// folded into Q A-fragments), online softmax in base-2, l via ones-WMMA,
// O = P Q.  Double-buffered LDS key tiles, 1 barrier / tile.
// ---------------------------------------------------------------------------
__global__ __launch_bounds__(256) void attn_kernel(const _Float16* __restrict__ qh,
                                                   _Float16* __restrict__ resh) {
  __shared__ __align__(16) _Float16 q_lds[2][KT][72];     // [key][d]  (pad 8)
  __shared__ __align__(16) _Float16 qt_lds[2][DH][40];    // [d][key]  (pad 8)
  __shared__ __align__(16) _Float16 p_lds[8][16][40];     // per-wave P tile

  const int lane = threadIdx.x & 31;
  const int wid  = threadIdx.x >> 5;
  const int bh   = blockIdx.y;                            // 0..31
  const _Float16* qbh = qh + (size_t)bh * NQ * DH;
  const int r0 = blockIdx.x * QT + wid * 16;              // this wave's rows

  // Q row fragments, pre-scaled by 8*log2(e) so exp() becomes raw v_exp_f32.
  v16h aq0 = load_a_frag(qbh + (size_t)r0 * DH + 0,  DH, lane);
  v16h aq1 = load_a_frag(qbh + (size_t)r0 * DH + 32, DH, lane);
  const _Float16 qscale = (_Float16)11.541561f;           // 8 * log2(e)
  aq0 = aq0 * qscale;
  aq1 = aq1 * qscale;

  v16h bones;                                             // ones B operand
#pragma unroll
  for (int i = 0; i < 16; ++i) bones[i] = (_Float16)1.0f;

  const v8f z = {};
  v8f acc0 = z, acc1 = z, acc2 = z, acc3 = z;             // O cols 0-15..48-63
  v8f mrow = z - 1e30f;                                   // running row max
  v8f lrow = z;                                           // running row sum

  // cooperative tile staging: 256 thr x 16B = 32x64 key tile
  const int crow = threadIdx.x >> 3;
  const int cch  = threadIdx.x & 7;

  v8h stage = *(const v8h*)(qbh + (size_t)crow * DH + cch * 8);
  *(v8h*)&q_lds[0][crow][cch * 8] = stage;
#pragma unroll
  for (int j = 0; j < 8; ++j) qt_lds[0][cch * 8 + j][crow] = stage[j];

  const int NT = NQ / KT;                                 // 64 tiles
  for (int it = 0; it < NT; ++it) {
    const int cur = it & 1;
    __syncthreads();                  // buf[cur] ready; buf[1-cur] consumed

    if (it + 1 < NT)                  // issue next tile load early
      stage = *(const v8h*)(qbh + (size_t)((it + 1) * KT + crow) * DH + cch * 8);
    if (it + 2 < NT)                  // global_prefetch_b8 for tile+2
      __builtin_prefetch(qbh + (size_t)((it + 2) * KT + crow) * DH + cch * 8, 0, 3);

    // ---- scores (log2 domain): S[n,c] = 8*log2e * sum_d Q[r0+n,d]Q[kt+c,d]
    const v16h b0k0 = load_b_frag(&q_lds[cur][0][0],   72, lane);
    const v16h b0k1 = load_b_frag(&q_lds[cur][0][32],  72, lane);
    const v16h b1k0 = load_b_frag(&q_lds[cur][16][0],  72, lane);
    const v16h b1k1 = load_b_frag(&q_lds[cur][16][32], 72, lane);
    const v8f s0 = wmma_f16(aq1, b0k1, wmma_f16(aq0, b0k0, z));
    const v8f s1 = wmma_f16(aq1, b1k1, wmma_f16(aq0, b1k0, z));

    // ---- online softmax (base 2) ----
    v8f t;
#pragma unroll
    for (int i = 0; i < 8; ++i) t[i] = fmaxf(s0[i], s1[i]);
    t = rowmax8(t);
    v8f mnew, corr;
#pragma unroll
    for (int i = 0; i < 8; ++i) {
      mnew[i] = fmaxf(mrow[i], t[i]);
      corr[i] = __builtin_amdgcn_exp2f(mrow[i] - mnew[i]);
    }
    mrow = mnew;
    lrow = lrow * corr;
    acc0 *= corr; acc1 *= corr; acc2 *= corr; acc3 *= corr;

    v8f p0, p1;
#pragma unroll
    for (int i = 0; i < 8; ++i) {
      p0[i] = __builtin_amdgcn_exp2f(s0[i] - mnew[i]);
      p1[i] = __builtin_amdgcn_exp2f(s1[i] - mnew[i]);
    }

    // ---- P: C-layout -> A-layout via per-wave LDS (f32 -> f16) ----
    {
      const int cb = lane & 15;
      const int rh = (lane >> 4) << 3;
#pragma unroll
      for (int i = 0; i < 8; ++i) {
        p_lds[wid][i + rh][cb]      = (_Float16)p0[i];
        p_lds[wid][i + rh][cb + 16] = (_Float16)p1[i];
      }
    }
    __builtin_amdgcn_wave_barrier();               // keep store->load order
    const v16h ap = load_a_frag(&p_lds[wid][0][0], 40, lane);

    // ---- row sums: l += P @ ones (one WMMA, replaces 32 ds_bpermute) ----
    lrow += wmma_f16(ap, bones, z);

    // ---- O += P[16x32] * Q_tile[32x64] (B[k=key][col=e] = qt_lds[e][key])
    const v16h bv0 = load_b_frag(&qt_lds[cur][0][0],  40, lane);
    const v16h bv1 = load_b_frag(&qt_lds[cur][16][0], 40, lane);
    const v16h bv2 = load_b_frag(&qt_lds[cur][32][0], 40, lane);
    const v16h bv3 = load_b_frag(&qt_lds[cur][48][0], 40, lane);
    acc0 = wmma_f16(ap, bv0, acc0);
    acc1 = wmma_f16(ap, bv1, acc1);
    acc2 = wmma_f16(ap, bv2, acc2);
    acc3 = wmma_f16(ap, bv3, acc3);

    // ---- stash next tile into the other buffer (readers already past) ----
    if (it + 1 < NT) {
      *(v8h*)&q_lds[1 - cur][crow][cch * 8] = stage;
#pragma unroll
      for (int j = 0; j < 8; ++j) qt_lds[1 - cur][cch * 8 + j][crow] = stage[j];
    }
  }

  // ---- normalize and scatter to res f16 [b*N+n][h*64+e] ----
  v8f inv;
#pragma unroll
  for (int i = 0; i < 8; ++i) inv[i] = __builtin_amdgcn_rcpf(lrow[i]);
  acc0 *= inv; acc1 *= inv; acc2 *= inv; acc3 *= inv;

  const int b  = bh >> 4, h = bh & 15;
  const int e0 = lane & 15;
  const int rh = (lane >> 4) << 3;
#pragma unroll
  for (int i = 0; i < 8; ++i) {
    const int n = r0 + i + rh;
    _Float16* o = resh + ((size_t)(b * NQ + n)) * DIM + h * DH + e0;
    o[0]  = (_Float16)acc0[i];
    o[16] = (_Float16)acc1[i];
    o[32] = (_Float16)acc2[i];
    o[48] = (_Float16)acc3[i];
  }
}

// ---------------------------------------------------------------------------
// Kernel 3: Y[4096,1024] = Res @ Wp^T + bp  (WMMA, direct-from-global frags;
// Wp_h row j IS the B-fragment [col][k] layout).
// ---------------------------------------------------------------------------
__global__ __launch_bounds__(256) void proj_kernel(const _Float16* __restrict__ resh,
                                                   const _Float16* __restrict__ wph,
                                                   const float* __restrict__ bp,
                                                   float* __restrict__ out) {
  const int lane = threadIdx.x & 31;
  const int wid  = threadIdx.x >> 5;
  const int r0 = blockIdx.x * 32 + (wid & 1) * 16;        // rows (token)
  const int j0 = blockIdx.y * 256 + (wid >> 1) * 64;      // out cols
  const v8f z = {};
  v8f c0 = z, c1 = z, c2 = z, c3 = z;

#pragma unroll 4
  for (int k = 0; k < DIM; k += 32) {
    const v16h a  = load_a_frag(resh + (size_t)r0 * DIM + k, DIM, lane);
    const v16h b0 = load_b_frag(wph + (size_t)(j0 +  0) * DIM + k, DIM, lane);
    const v16h b1 = load_b_frag(wph + (size_t)(j0 + 16) * DIM + k, DIM, lane);
    const v16h b2 = load_b_frag(wph + (size_t)(j0 + 32) * DIM + k, DIM, lane);
    const v16h b3 = load_b_frag(wph + (size_t)(j0 + 48) * DIM + k, DIM, lane);
    c0 = wmma_f16(a, b0, c0);
    c1 = wmma_f16(a, b1, c1);
    c2 = wmma_f16(a, b2, c2);
    c3 = wmma_f16(a, b3, c3);
  }

  const int jl = lane & 15;
  const int rh = (lane >> 4) << 3;
  const float b0 = bp[j0 + jl], b1 = bp[j0 + 16 + jl];
  const float b2 = bp[j0 + 32 + jl], b3 = bp[j0 + 48 + jl];
#pragma unroll
  for (int i = 0; i < 8; ++i) {
    float* o = out + (size_t)(r0 + rh + i) * DIM + j0 + jl;
    o[0]  = c0[i] + b0;
    o[16] = c1[i] + b1;
    o[32] = c2[i] + b2;
    o[48] = c3[i] + b3;
  }
}

// ---------------------------------------------------------------------------
extern "C" void kernel_launch(void* const* d_in, const int* in_sizes, int n_in,
                              void* d_out, int out_size, void* d_ws, size_t ws_size,
                              hipStream_t stream) {
  const float* x  = (const float*)d_in[0];   // [2,2048,1024]
  const float* Wq = (const float*)d_in[1];   // [16,64,64]
  const float* Wp = (const float*)d_in[2];   // [1024,1024]
  const float* bp = (const float*)d_in[3];   // [1024]
  float* out = (float*)d_out;                // [2,2048,1024] f32

  char* ws = (char*)d_ws;
  _Float16* qh   = (_Float16*)(ws);                       // 8 MB  [BH][N][64]
  _Float16* resh = (_Float16*)(ws + (size_t)(8u  << 20)); // 8 MB  [B*N][1024]
  _Float16* wph  = (_Float16*)(ws + (size_t)(16u << 20)); // 2 MB  [1024][1024]

  wpconv_kernel<<<dim3(1024), 256, 0, stream>>>(Wp, wph);
  qproj_kernel <<<dim3(NQ / 4, BSZ * NH), 256, 0, stream>>>(x, Wq, qh);
  attn_kernel  <<<dim3(NQ / QT, BSZ * NH), 256, 0, stream>>>(qh, resh);
  proj_kernel  <<<dim3(BSZ * NQ / 32, DIM / 256), 256, 0, stream>>>(resh, wph, bp, out);
}
